// Attention_43963285242627
// MI455X (gfx1250) — matrix operand order repaired and checked by reading, not compiled, not run on hardware
//
#include <hip/hip_runtime.h>
#include <hip/hip_bf16.h>

// ---------------------------------------------------------------------------
// Problem constants (from reference): B=2, L=2048, D=2048, H=32, HKV=8, HD=64
// ---------------------------------------------------------------------------
#define BATCH 2
#define SEQ   2048
#define DMODEL 2048
#define NH    32
#define NKV   8
#define HD    64
#define GQA   (NH / NKV)          // 4
#define EPS_  1e-5f
// log2(1e6)/32  (RoPE inv_freq = theta^(-i/32) = 2^(-i*log2(theta)/32))
#define ROPE_L2 0.62286151789911680f

typedef __attribute__((ext_vector_type(16))) __bf16 v16bf;
typedef __attribute__((ext_vector_type(8)))  float  v8f;

// One WMMA operand = 16 bf16 per lane = 32 contiguous bytes.
union Frag {
    v16bf v;
    unsigned short u[16];
    unsigned int   w[8];
    uint4          q2[2];
};

// Native f32 -> bf16 conversion (hardware cvt, not manual bit-twiddling).
__device__ __forceinline__ unsigned short f2bf(float f) {
    union { __bf16 b; unsigned short s; } u;
    u.b = (__bf16)f;
    return u.s;
}
__device__ __forceinline__ unsigned int f2bf2(float lo, float hi) {
    return (unsigned int)f2bf(lo) | ((unsigned int)f2bf(hi) << 16);
}

// ---------------------------------------------------------------------------
// GEMM: C[M,N] = A[M,K] (fp32) @ B[K,N] (fp32), bf16 WMMA accumulate fp32.
// Block: 256 threads = 8 waves in 4(M) x 2(N) arrangement of 32x32 wave tiles.
// Tile: BM=128, BN=64, BK=32.  LDS holds tiles ALREADY PERMUTED into WMMA
// fragment order ([tile][lane][16 elems], 32 B/lane) so each operand is one
// v16bf load (2 x ds_load_b128).
//
// A-layout (16-bit A 16x32): lane = (row&15) + 16*((k>>3)&1),
//                            elem = (k&7) + 8*((k>>4)&1)
// B-layout (16-bit B 32x16): lane = (n&15) + 16*((k>>3)&1),
//                            elem = (k&7) + 8*((k>>4)&1)
// ---------------------------------------------------------------------------
#define BM 128
#define BN 64
#define BK 32

__global__ __launch_bounds__(256)
void gemm_bf16_wmma(const float* __restrict__ A, const float* __restrict__ Bm,
                    float* __restrict__ C, int M, int N, int K)
{
    __shared__ __attribute__((aligned(32))) unsigned short Asf[8][32][16]; // 8 KB
    __shared__ __attribute__((aligned(32))) unsigned short Bsf[4][32][16]; // 4 KB

    const int tid  = threadIdx.x;
    const int lane = tid & 31;
    const int wave = tid >> 5;
    const int wm = wave >> 1;      // 0..3
    const int wn = wave & 1;       // 0..1
    const int bm = blockIdx.y * BM;
    const int bn = blockIdx.x * BN;
    const int grp = lane >> 4;
    const int col = lane & 15;

    v8f acc[2][2] = {};

    for (int kk = 0; kk < K; kk += BK) {
        // ---- prefetch next K-step tiles (lowers to global_prefetch_b8) ----
        if (kk + BK < K) {
            if (tid < 128) {
                // next A tile: 128 rows x 128 B
                __builtin_prefetch(A + (size_t)(bm + tid) * K + kk + BK, 0, 0);
            } else {
                // next B rows: 64 rows x 256 B -> 128 cache lines
                int r    = (tid - 128) >> 1;
                int half = tid & 1;
                __builtin_prefetch(Bm + (size_t)(kk + BK + r) * N + bn + half * 32,
                                   0, 0);
            }
        }

        // ---- stage A 128x32: float2 coalesced loads, fragment-order stores --
        #pragma unroll
        for (int i = 0; i < 8; ++i) {
            int idx = tid + i * 256;            // pair index
            int r  = idx >> 4;                  // 0..127
            int c2 = idx & 15;                  // pair within row
            int c  = c2 * 2;                    // 0..30 (even)
            float2 a2 = ((const float2*)(A + (size_t)(bm + r) * K + kk))[c2];
            int mtile    = r >> 4;
            int lane_dst = (r & 15) + 16 * ((c >> 3) & 1);
            int e        = (c & 7) + 8 * ((c >> 4) & 1);   // even
            ((unsigned int*)&Asf[mtile][lane_dst][0])[e >> 1] = f2bf2(a2.x, a2.y);
        }
        // ---- stage B 32x64: two coalesced row reads, fragment-order stores --
        #pragma unroll
        for (int i = 0; i < 4; ++i) {
            int idx = tid + i * 256;            // pair index (pairs along k)
            int n  = idx & 63;
            int k  = (idx >> 6) * 2;            // 0..30 (even)
            float b0 = Bm[(size_t)(kk + k)     * N + bn + n];
            float b1 = Bm[(size_t)(kk + k + 1) * N + bn + n];
            int ntile    = n >> 4;
            int lane_dst = (n & 15) + 16 * ((k >> 3) & 1);
            ((unsigned int*)&Bsf[ntile][lane_dst][0])[((k & 7) + 8 * ((k >> 4) & 1)) >> 1] = f2bf2(b0, b1);
        }
        __syncthreads();

        Frag aF[2], bF[2];
        #pragma unroll
        for (int t = 0; t < 2; ++t) {
            aF[t].v = *(const v16bf*)&Asf[wm * 2 + t][lane][0];
            bF[t].v = *(const v16bf*)&Bsf[wn * 2 + t][lane][0];
        }

        #pragma unroll
        for (int i = 0; i < 2; ++i)
            #pragma unroll
            for (int j = 0; j < 2; ++j)
                acc[i][j] = __builtin_amdgcn_wmma_f32_16x16x32_bf16(
                    false, aF[i].v, false, bF[j].v, (short)0, acc[i][j],
                    false, false);
        __syncthreads();
    }

    // ---- store (32-bit offsets -> SGPR base + VGPR offset addressing) ----
    #pragma unroll
    for (int i = 0; i < 2; ++i)
        #pragma unroll
        for (int j = 0; j < 2; ++j)
            #pragma unroll
            for (int r = 0; r < 8; ++r) {
                int row = bm + wm * 32 + i * 16 + 8 * grp + r;
                int off = row * N + bn + wn * 32 + j * 16 + col;
                C[off] = acc[i][j][r];
            }
}

// ---------------------------------------------------------------------------
// Fused RMS-norm + RoPE + head transpose + fp32->bf16.
// grid: (B*L, H + 2*HKV); block: 64 threads (one head vector).
// Outputs: qb[B][H][L][HD], kb[B][HKV][L][HD]  (bf16, head-dim contiguous)
//          vbT[B][HKV][HD][L]                  (bf16, TRANSPOSED: contraction
//                                               dim (seq) contiguous for P@V)
// ---------------------------------------------------------------------------
__global__ __launch_bounds__(64)
void rope_qkv(const float* __restrict__ q, const float* __restrict__ k,
              const float* __restrict__ v, const float* __restrict__ qw,
              const float* __restrict__ kw,
              unsigned short* __restrict__ qb, unsigned short* __restrict__ kb,
              unsigned short* __restrict__ vbT)
{
    __shared__ float red[64];
    __shared__ float sh[64];

    const int bl = blockIdx.x;          // b*L + l
    const int b  = bl >> 11;
    const int l  = bl & (SEQ - 1);
    const int y  = blockIdx.y;
    const int d  = threadIdx.x;

    if (y >= NH + NKV) {                // ---- V: transpose + convert only ----
        int hv = y - NH - NKV;
        float val = v[(size_t)bl * (NKV * HD) + hv * HD + d];
        vbT[((size_t)(b * NKV + hv) * HD + d) * SEQ + l] = f2bf(val);
        return;
    }

    const float* src;
    const float* w;
    unsigned short* dst;
    if (y < NH) {
        src = q + (size_t)bl * (NH * HD) + y * HD;
        w = qw;
        dst = qb + (((size_t)(b * NH + y)) * SEQ + l) * HD;
    } else {
        int hk = y - NH;
        src = k + (size_t)bl * (NKV * HD) + hk * HD;
        w = kw;
        dst = kb + (((size_t)(b * NKV + hk)) * SEQ + l) * HD;
    }

    float val = src[d];

    // RMS norm over HD=64
    red[d] = val * val;
    __syncthreads();
    #pragma unroll
    for (int s = 32; s > 0; s >>= 1) {
        if (d < s) red[d] += red[d + s];
        __syncthreads();
    }
    float rms = __frsqrt_rn(red[0] * (1.0f / HD) + EPS_);
    val = val * rms * w[d];

    // RoPE (half-split form)
    sh[d] = val;
    __syncthreads();
    float partner = sh[d ^ 32];
    int i = d & 31;
    float inv = exp2f(-ROPE_L2 * (float)i);
    float ang = (float)l * inv;
    float sn, cs;
    __sincosf(ang, &sn, &cs);
    val = (d < 32) ? (val * cs - partner * sn)
                   : (partner * cs + val * sn);
    dst[d] = f2bf(val);
}

// ---------------------------------------------------------------------------
// Flash attention (causal, GQA 4:1). One wave per 16-row q tile.
// grid: (L/16, B*H); block: 32 threads (wave32, EXEC all-ones for WMMA).
//
// WMMA operands are read as 16 B runs (P: one 32 B run):
//   Q  : [head][seq][hd]  -> A-frag = 2 x 16 B global runs   (kept in VGPRs)
//   K  : [head][seq][hd]  -> B-frag = 2 x 16 B global runs (contraction over hd)
//   V^T: [head][hd][seq]  -> B-frag = 2 x 16 B global runs (contraction over seq)
//   P  : staged in LDS in A-fragment order -> 1 v16bf read per chunk
// ---------------------------------------------------------------------------
__global__ __launch_bounds__(32)
void flash_attn(const unsigned short* __restrict__ qb,
                const unsigned short* __restrict__ kb,
                const unsigned short* __restrict__ vbT,
                float* __restrict__ attn)
{
    __shared__ __attribute__((aligned(32))) unsigned short Psf[32][16]; // 1 KB

    const int lane = threadIdx.x;
    const int qi = blockIdx.x;
    const int bh = blockIdx.y;
    const int b = bh / NH;
    const int h = bh % NH;
    const int hk = h / GQA;

    const int grp = lane >> 4;
    const int col = lane & 15;
    const int rowL = lane & 15;

    const unsigned short* Qp  = qb  + (((size_t)(b * NH + h)) * SEQ + qi * 16) * HD;
    const unsigned short* Kp  = kb  + ((size_t)(b * NKV + hk)) * SEQ * HD;
    const unsigned short* VpT = vbT + ((size_t)(b * NKV + hk)) * HD * SEQ;

    // Q fragments: two K=32 A-fragments over HD=64, each = 2 x 16 B runs
    Frag aQ[2];
    const unsigned short* qrow = Qp + rowL * HD;
    #pragma unroll
    for (int t = 0; t < 2; ++t) {
        aQ[t].q2[0] = *(const uint4*)(qrow + t * 32 + 8 * grp);
        aQ[t].q2[1] = *(const uint4*)(qrow + t * 32 + 16 + 8 * grp);
    }

    v8f accO[4] = {};
    float m[8], lsum[8];
    #pragma unroll
    for (int r = 0; r < 8; ++r) { m[r] = -3.0e38f; lsum[r] = 0.0f; }

    const int qmax = qi * 16 + 15;
    const float scl = 0.125f;          // HD^-0.5
    const int gsel = (col >> 3) & 1;   // P-store fragment lane select
    const int e0   = col & 7;

    for (int kc = 0; kc <= qmax; kc += 32) {
        // ---- S = Q K^T: B-fragments straight from global (2 x 16 B runs) ----
        v8f sacc[2] = {};
        #pragma unroll
        for (int dc = 0; dc < 2; ++dc) {
            #pragma unroll
            for (int kt = 0; kt < 2; ++kt) {
                Frag bK;
                bK.q2[0] = *(const uint4*)(Kp + (size_t)(kc + kt * 16 + col) * HD
                                          + dc * 32 + 8 * grp);
                bK.q2[1] = *(const uint4*)(Kp + (size_t)(kc + kt * 16 + col) * HD
                                          + dc * 32 + 16 + 8 * grp);
                sacc[kt] = __builtin_amdgcn_wmma_f32_16x16x32_bf16(
                    false, aQ[dc].v, false, bK.v, (short)0, sacc[kt],
                    false, false);
            }
        }

        const bool need_mask = (kc + 31 > qmax);

        // ---- online softmax (C/D layout: VGPR r -> row r + 8*grp) ----
        #pragma unroll
        for (int r = 0; r < 8; ++r) {
            int row = qi * 16 + r + 8 * grp;
            float s0 = sacc[0][r] * scl;
            float s1 = sacc[1][r] * scl;
            if (need_mask) {
                if (kc + col      > row) s0 = -3.0e38f;
                if (kc + 16 + col > row) s1 = -3.0e38f;
            }
            float mx = fmaxf(s0, s1);
            #pragma unroll
            for (int off = 1; off < 16; off <<= 1)
                mx = fmaxf(mx, __shfl_xor(mx, off, 16));

            float mnew = fmaxf(m[r], mx);
            float corr = __expf(m[r] - mnew);
            float p0 = __expf(s0 - mnew);
            float p1 = __expf(s1 - mnew);
            float rs = p0 + p1;
            #pragma unroll
            for (int off = 1; off < 16; off <<= 1)
                rs += __shfl_xor(rs, off, 16);

            lsum[r] = lsum[r] * corr + rs;
            m[r] = mnew;
            #pragma unroll
            for (int t = 0; t < 4; ++t)
                accO[t][r] = accO[t][r] * corr;

            // store P directly in A-fragment order:
            //   frag lane = row_in_tile + 16*((key>>3)&1)
            //   frag elem = (key&7) + 8*(key>>4)
            int lane_r = (r + 8 * grp) + 16 * gsel;
            Psf[lane_r][e0]     = f2bf(p0);   // key = kc + col       (tile 0)
            Psf[lane_r][e0 + 8] = f2bf(p1);   // key = kc + 16 + col  (tile 1)
        }
        __syncthreads();

        // ---- O += P V : P A-frag is one contiguous 32 B LDS read ----
        Frag aP;
        aP.v = *(const v16bf*)&Psf[lane][0];
        #pragma unroll
        for (int t = 0; t < 4; ++t) {
            Frag bV;
            bV.q2[0] = *(const uint4*)(VpT + (size_t)(t * 16 + col) * SEQ
                                       + kc + 8 * grp);
            bV.q2[1] = *(const uint4*)(VpT + (size_t)(t * 16 + col) * SEQ
                                       + kc + 16 + 8 * grp);
            accO[t] = __builtin_amdgcn_wmma_f32_16x16x32_bf16(
                false, aP.v, false, bV.v, (short)0, accO[t], false, false);
        }
        __syncthreads();
    }

    // ---- epilogue: normalize, scatter to attn[B, L, H*HD] (fp32) ----
    #pragma unroll
    for (int t = 0; t < 4; ++t)
        #pragma unroll
        for (int r = 0; r < 8; ++r) {
            int row = qi * 16 + r + 8 * grp;
            int off = (b * SEQ + row) * (NH * HD) + h * HD + t * 16 + col;
            attn[off] = accO[t][r] / lsum[r];
        }
}

// ---------------------------------------------------------------------------
// Launch: QKV GEMMs -> rope/norm/convert -> flash attention -> Wo GEMM
// ---------------------------------------------------------------------------
extern "C" void kernel_launch(void* const* d_in, const int* in_sizes, int n_in,
                              void* d_out, int out_size, void* d_ws, size_t ws_size,
                              hipStream_t stream)
{
    const float* x   = (const float*)d_in[0];
    const float* Wq  = (const float*)d_in[1];
    const float* Wk  = (const float*)d_in[2];
    const float* Wv  = (const float*)d_in[3];
    const float* Wo  = (const float*)d_in[4];
    const float* qnw = (const float*)d_in[5];
    const float* knw = (const float*)d_in[6];

    const size_t QN = (size_t)BATCH * SEQ * NH * HD;   // 8,388,608
    const size_t KN = (size_t)BATCH * SEQ * NKV * HD;  // 2,097,152

    float* q = (float*)d_ws;                 // fp32 q, later reused as attn out
    float* k = q + QN;
    float* v = k + KN;
    unsigned short* qb  = (unsigned short*)(v + KN);
    unsigned short* kb  = qb + QN;
    unsigned short* vbT = kb + KN;
    float* attn = q;                         // alias: fp32 q dead after rope

    const int M = BATCH * SEQ;               // 4096

    // 1-3: QKV projections
    gemm_bf16_wmma<<<dim3((NH * HD) / BN, M / BM), 256, 0, stream>>>(
        x, Wq, q, M, NH * HD, DMODEL);
    gemm_bf16_wmma<<<dim3((NKV * HD) / BN, M / BM), 256, 0, stream>>>(
        x, Wk, k, M, NKV * HD, DMODEL);
    gemm_bf16_wmma<<<dim3((NKV * HD) / BN, M / BM), 256, 0, stream>>>(
        x, Wv, v, M, NKV * HD, DMODEL);

    // 4: RMS-norm + RoPE + transpose + bf16 convert (V transposed)
    rope_qkv<<<dim3(BATCH * SEQ, NH + 2 * NKV), 64, 0, stream>>>(
        q, k, v, qnw, knw, qb, kb, vbT);

    // 5: flash attention (writes fp32 into attn == q alias)
    flash_attn<<<dim3(SEQ / 16, BATCH * NH), 32, 0, stream>>>(qb, kb, vbT, attn);

    // 6: output projection
    gemm_bf16_wmma<<<dim3(DMODEL / BN, M / BM), 256, 0, stream>>>(
        attn, Wo, (float*)d_out, M, DMODEL, DMODEL);
}
